// HeteroLinkPredictionModel_64553358459086
// MI455X (gfx1250) — matrix operand order repaired and checked by reading, not compiled, not run on hardware
//
#include <hip/hip_runtime.h>
#include <hip/hip_bf16.h>

// ---------------- model dimensions ----------------
#define NP   50000
#define NC   5000
#define NE   200000
#define NEL  100000
#define DP   1280
#define DC   256
#define HID  128
#define HEADS 4

typedef __bf16 bf16_t;
typedef __attribute__((ext_vector_type(16))) __bf16 v16bf;
typedef __attribute__((ext_vector_type(8)))  __bf16 v8bf;
typedef __attribute__((ext_vector_type(8)))  float  v8f;

// round-to-nearest-even f32 -> bf16, two at a time, packed into one u32
// (low halfword = first element => memory order matches bf16 stream)
__device__ __forceinline__ unsigned pack2bf(float f0, float f1) {
  unsigned u0 = __builtin_bit_cast(unsigned, f0);
  unsigned u1 = __builtin_bit_cast(unsigned, f1);
  unsigned r0 = (u0 + 0x7FFFu + ((u0 >> 16) & 1u)) >> 16;
  unsigned r1 = (u1 + 0x7FFFu + ((u1 >> 16) & 1u)) >> 16;
  return (r1 << 16) | (r0 & 0xFFFFu);
}

// order-preserving float <-> uint encoding for atomicMax on floats
__device__ __forceinline__ unsigned fenc(float f) {
  unsigned u = __builtin_bit_cast(unsigned, f);
  return (u & 0x80000000u) ? ~u : (u | 0x80000000u);
}
__device__ __forceinline__ float fdec(unsigned u) {
  unsigned b = (u & 0x80000000u) ? (u & 0x7FFFFFFFu) : ~u;
  return __builtin_bit_cast(float, b);
}

// =====================================================================
// WMMA GEMM:  D[M,N] = act( A[M,K] @ B[K,N] (+Cin) (+bias) )
// block = 256 thr = 8 waves; block tile = 32 (M) x 128 (N); K step 32.
// Each wave owns a 16-column strip and computes BOTH 16-row subtiles,
// reusing one B fragment for two WMMAs per K step.
// LDS staged as packed-bf16 u32 words; fragment reads are 16B ds_load_b128.
// GUARD=false: block tile guaranteed fully inside [0,M): branch-free.
// FLAGS: bit0 = accumulate Cin, bit1 = add bias, bit2 = ReLU
// =====================================================================
template <int FLAGS, bool GUARD>
__global__ __launch_bounds__(256) void gemm_bf16_wmma(
    const float* __restrict__ A, const float* __restrict__ B,
    const float* __restrict__ Cin, const float* __restrict__ bias,
    float* __restrict__ D, int m_base, int M, int N, int K)
{
  // row strides of 20 u32 (80B) keep 16B alignment and spread banks
  __shared__ __align__(16) unsigned sA32[32 * 20];   // A tile 32x32 bf16 (packed)
  __shared__ __align__(16) unsigned sB32[128 * 20];  // B^T tile 128x32 bf16 (packed)

  const int tid  = threadIdx.x;
  const int wave = tid >> 5;
  const int lane = tid & 31;
  const int m0   = m_base + blockIdx.x * 32;
  const int n0   = blockIdx.y * 128;
  const int mloc = lane & 15;
  const int half = lane >> 4;
  const int ncol = n0 + wave * 16 + mloc;           // this lane's C column

  // per-thread staging coordinates
  const int ar  = (tid * 2) >> 4;                   // A row 0..31 (two pairs/thread)
  const int acp = (tid * 2) & 15;                   // A column-pair base
  const int bkp = (tid * 8) >> 7;                   // B k-pair 0..15
  const int bnn = (tid * 8) & 127;                  // B col base 0..120

  v8f acc0 = {0.f, 0.f, 0.f, 0.f, 0.f, 0.f, 0.f, 0.f};
  v8f acc1 = {0.f, 0.f, 0.f, 0.f, 0.f, 0.f, 0.f, 0.f};

  for (int k0 = 0; k0 < K; k0 += 32) {
    // ---- stage A tile 32x32: 512 u32, 2 per thread ----
    {
      int gr = m0 + ar;
      const float* ap = A + (size_t)gr * K + k0 + 2 * acp;
      if (!GUARD || gr < M) {
        sA32[ar * 20 + acp]     = pack2bf(ap[0], ap[1]);
        sA32[ar * 20 + acp + 1] = pack2bf(ap[2], ap[3]);
        if (k0 + 32 < K) __builtin_prefetch(ap + 32, 0, 0);   // next K tile
      } else {
        sA32[ar * 20 + acp]     = 0u;
        sA32[ar * 20 + acp + 1] = 0u;
      }
    }
    // ---- stage B tile 32x128 transposed: 2048 u32, 8 per thread ----
    {
      const float* b0row = B + (size_t)(k0 + 2 * bkp) * N + n0 + bnn;
      const float* b1row = b0row + N;
#pragma unroll
      for (int i = 0; i < 8; ++i)
        sB32[(bnn + i) * 20 + bkp] = pack2bf(b0row[i], b1row[i]);
      if (k0 + 32 < K) __builtin_prefetch(b0row + (size_t)32 * N, 0, 0);
    }
    __syncthreads();

    // ---- fragments: contiguous 16B loads per ISA 16-bit layouts ----
    const v8bf b0 = *(const v8bf*)&sB32[(wave * 16 + mloc) * 20 + half * 4];
    const v8bf b1 = *(const v8bf*)&sB32[(wave * 16 + mloc) * 20 + 8 + half * 4];
    const v8bf a00 = *(const v8bf*)&sA32[mloc * 20 + half * 4];
    const v8bf a01 = *(const v8bf*)&sA32[mloc * 20 + 8 + half * 4];
    const v8bf a10 = *(const v8bf*)&sA32[(16 + mloc) * 20 + half * 4];
    const v8bf a11 = *(const v8bf*)&sA32[(16 + mloc) * 20 + 8 + half * 4];
    v16bf a0, a1, b;
#pragma unroll
    for (int i = 0; i < 8; ++i) {
      b[i]  = b0[i];  b[8 + i]  = b1[i];
      a0[i] = a00[i]; a0[8 + i] = a01[i];
      a1[i] = a10[i]; a1[8 + i] = a11[i];
    }

    acc0 = __builtin_amdgcn_wmma_f32_16x16x32_bf16(
        false, a0, false, b, (short)0, acc0, false, false);
    acc1 = __builtin_amdgcn_wmma_f32_16x16x32_bf16(
        false, a1, false, b, (short)0, acc1, false, false);
    __syncthreads();
  }

  // ---- epilogue: C/D layout -> row m0 + sub*16 + half*8 + r, col ncol ----
  if (!GUARD) {                           // branch-free path
#pragma unroll
    for (int r = 0; r < 8; ++r) {
      int gm0 = m0 + half * 8 + r;
      int gm1 = gm0 + 16;
      float v0 = acc0[r], v1 = acc1[r];
      if (FLAGS & 1) { v0 += Cin[(size_t)gm0 * N + ncol]; v1 += Cin[(size_t)gm1 * N + ncol]; }
      if (FLAGS & 2) { float bb = bias[ncol]; v0 += bb; v1 += bb; }
      if (FLAGS & 4) { v0 = fmaxf(v0, 0.f); v1 = fmaxf(v1, 0.f); }
      D[(size_t)gm0 * N + ncol] = v0;
      D[(size_t)gm1 * N + ncol] = v1;
    }
  } else {
#pragma unroll
    for (int s2 = 0; s2 < 2; ++s2) {
      v8f acc = s2 ? acc1 : acc0;
#pragma unroll
      for (int r = 0; r < 8; ++r) {
        int gm = m0 + s2 * 16 + half * 8 + r;
        if (gm < M) {
          float v = acc[r];
          if (FLAGS & 1) v += Cin[(size_t)gm * N + ncol];
          if (FLAGS & 2) v += bias[ncol];
          if (FLAGS & 4) v = fmaxf(v, 0.f);
          D[(size_t)gm * N + ncol] = v;
        }
      }
    }
  }
}

// ================= SAGE scatter-mean (projected-first) =================
__global__ __launch_bounds__(256) void sage_scatter_kernel(
    const float* __restrict__ y, const int* __restrict__ isrc,
    const int* __restrict__ idst, float* __restrict__ agg,
    float* __restrict__ cnt, int E)
{
  int wave = threadIdx.x >> 5, lane = threadIdx.x & 31;
  int e = blockIdx.x * 8 + wave;
  if (e >= E) return;
  int s = isrc[e], d = idst[e];
  const float* r = y + (size_t)s * HID;
  float* a = agg + (size_t)d * HID;
#pragma unroll
  for (int i = 0; i < 4; ++i)
    atomicAdd(&a[lane + 32 * i], r[lane + 32 * i]);
  if (lane == 0) atomicAdd(&cnt[d], 1.0f);
}

__global__ void div_cnt_kernel(float* __restrict__ agg,
                               const float* __restrict__ cnt, int Nn)
{
  int idx = blockIdx.x * blockDim.x + threadIdx.x;
  if (idx >= Nn * HID) return;
  agg[idx] /= fmaxf(cnt[idx >> 7], 1.0f);
}

// ============== GAT: per-node attention logits  (nh = hs . a_h) ==============
__global__ __launch_bounds__(256) void attn_logits_kernel(
    const float* __restrict__ hfeat, const float* __restrict__ avec,
    float* __restrict__ out, int Nn)
{
  int wave = threadIdx.x >> 5, lane = threadIdx.x & 31;
  int n = blockIdx.x * 8 + wave;
  if (n >= Nn) return;
  const float* row = hfeat + (size_t)n * (HEADS * HID);
#pragma unroll
  for (int h = 0; h < HEADS; ++h) {
    float s = 0.f;
#pragma unroll
    for (int i = 0; i < 4; ++i) {
      int c = lane + 32 * i;
      s += row[h * HID + c] * avec[h * HID + c];
    }
    for (int off = 16; off > 0; off >>= 1) s += __shfl_down(s, off, 32);
    if (lane == 0) out[n * HEADS + h] = s;
  }
}

__global__ void fill_u32_kernel(unsigned* __restrict__ p, unsigned v, int n)
{
  int i = blockIdx.x * blockDim.x + threadIdx.x;
  if (i < n) p[i] = v;
}

// pass 1: segment max (stable softmax)
__global__ void edge_max_kernel(
    const float* __restrict__ als, const float* __restrict__ ald,
    const int* __restrict__ is, const int* __restrict__ id,
    unsigned* __restrict__ emax, int E)
{
  int e = blockIdx.x * blockDim.x + threadIdx.x;
  if (e >= E) return;
  int s = is[e], d = id[e];
#pragma unroll
  for (int h = 0; h < HEADS; ++h) {
    float x = als[s * HEADS + h] + ald[d * HEADS + h];
    x = (x > 0.f) ? x : 0.2f * x;                 // leaky_relu 0.2
    atomicMax(&emax[d * HEADS + h], fenc(x));
  }
}

// pass 2: softmax denominators
__global__ void edge_den_kernel(
    const float* __restrict__ als, const float* __restrict__ ald,
    const int* __restrict__ is, const int* __restrict__ id,
    const unsigned* __restrict__ emax, float* __restrict__ den, int E)
{
  int e = blockIdx.x * blockDim.x + threadIdx.x;
  if (e >= E) return;
  int s = is[e], d = id[e];
#pragma unroll
  for (int h = 0; h < HEADS; ++h) {
    float x = als[s * HEADS + h] + ald[d * HEADS + h];
    x = (x > 0.f) ? x : 0.2f * x;
    atomicAdd(&den[d * HEADS + h], __expf(x - fdec(emax[d * HEADS + h])));
  }
}

// pass 3: weighted scatter  att[d] += alpha_h * hs[s]   (512-wide, wave/edge)
__global__ __launch_bounds__(256) void edge_scatter_kernel(
    const float* __restrict__ als, const float* __restrict__ ald,
    const unsigned* __restrict__ emax, const float* __restrict__ den,
    const float* __restrict__ hs, const int* __restrict__ is,
    const int* __restrict__ id, float* __restrict__ att, int E)
{
  int wave = threadIdx.x >> 5, lane = threadIdx.x & 31;
  int e = blockIdx.x * 8 + wave;
  if (e >= E) return;
  int s = is[e], d = id[e];
  float alpha[HEADS];
#pragma unroll
  for (int h = 0; h < HEADS; ++h) {
    float x = als[s * HEADS + h] + ald[d * HEADS + h];
    x = (x > 0.f) ? x : 0.2f * x;
    alpha[h] = __expf(x - fdec(emax[d * HEADS + h])) / den[d * HEADS + h];
  }
  const float* hrow = hs + (size_t)s * (HEADS * HID);
  float* arow = att + (size_t)d * (HEADS * HID);
#pragma unroll
  for (int i = 0; i < 16; ++i) {
    int c = lane * 16 + i;
    atomicAdd(&arow[c], alpha[c >> 7] * hrow[c]);
  }
}

// mean over heads + bias + relu
__global__ void head_mean_kernel(const float* __restrict__ att,
                                 const float* __restrict__ b,
                                 float* __restrict__ g, int Nn)
{
  int idx = blockIdx.x * blockDim.x + threadIdx.x;
  if (idx >= Nn * HID) return;
  int n = idx >> 7, c = idx & 127;
  const float* r = att + (size_t)n * (HEADS * HID);
  float v = 0.25f * (r[c] + r[HID + c] + r[2 * HID + c] + r[3 * HID + c]) + b[c];
  g[idx] = fmaxf(v, 0.f);
}

// collapse lin_{p,c} @ link_W into per-node dot vectors + scalar constant
__global__ void combine_vec_kernel(
    const float* __restrict__ linpW, const float* __restrict__ linpB,
    const float* __restrict__ lincW, const float* __restrict__ lincB,
    const float* __restrict__ linkW, const float* __restrict__ linkB,
    float* __restrict__ vp, float* __restrict__ vc, float* __restrict__ s0)
{
  int c = threadIdx.x;   // 0..127
  float ap = 0.f, ac = 0.f;
  for (int o = 0; o < HID; ++o) {
    ap += linpW[c * HID + o] * linkW[o];
    ac += lincW[c * HID + o] * linkW[HID + o];
  }
  vp[c] = ap; vc[c] = ac;
  __shared__ float red[HID];
  red[c] = linpB[c] * linkW[c] + lincB[c] * linkW[HID + c];
  __syncthreads();
  if (c == 0) {
    float t = linkB[0];
    for (int i = 0; i < HID; ++i) t += red[i];
    *s0 = t;
  }
}

__global__ __launch_bounds__(256) void node_score_kernel(
    const float* __restrict__ g, const float* __restrict__ v,
    float* __restrict__ out, int Nn)
{
  int wave = threadIdx.x >> 5, lane = threadIdx.x & 31;
  int n = blockIdx.x * 8 + wave;
  if (n >= Nn) return;
  const float* r = g + (size_t)n * HID;
  float s = 0.f;
#pragma unroll
  for (int i = 0; i < 4; ++i) s += r[lane + 32 * i] * v[lane + 32 * i];
  for (int off = 16; off > 0; off >>= 1) s += __shfl_down(s, off, 32);
  if (lane == 0) out[n] = s;
}

__global__ void link_kernel(const float* __restrict__ sp,
                            const float* __restrict__ sc,
                            const float* __restrict__ s0,
                            const int* __restrict__ es,
                            const int* __restrict__ ed,
                            float* __restrict__ out, int EL)
{
  int e = blockIdx.x * blockDim.x + threadIdx.x;
  if (e >= EL) return;
  float z = sp[es[e]] + sc[ed[e]] + *s0;
  out[e] = 1.f / (1.f + __expf(-z));
}

// =====================================================================
static void launch_gemm(int flags, const float* A, const float* B,
                        const float* Cin, const float* bias, float* D,
                        int M, int N, int K, hipStream_t s)
{
  int mfull = M / 32;                 // fully-covered 32-row blocks
  int mtail = M - mfull * 32;
  if (mfull > 0) {
    dim3 grid(mfull, N / 128);
    if (flags == 0)
      gemm_bf16_wmma<0, false><<<grid, 256, 0, s>>>(A, B, Cin, bias, D, 0, M, N, K);
    else
      gemm_bf16_wmma<7, false><<<grid, 256, 0, s>>>(A, B, Cin, bias, D, 0, M, N, K);
  }
  if (mtail > 0) {
    dim3 grid(1, N / 128);
    if (flags == 0)
      gemm_bf16_wmma<0, true><<<grid, 256, 0, s>>>(A, B, Cin, bias, D, mfull * 32, M, N, K);
    else
      gemm_bf16_wmma<7, true><<<grid, 256, 0, s>>>(A, B, Cin, bias, D, mfull * 32, M, N, K);
  }
}

static float* bump(char*& p, size_t nfloats)
{
  float* r = (float*)p;
  p += ((nfloats * sizeof(float) + 255) & ~(size_t)255);
  return r;
}

extern "C" void kernel_launch(void* const* d_in, const int* in_sizes, int n_in,
                              void* d_out, int out_size, void* d_ws, size_t ws_size,
                              hipStream_t stream)
{
  (void)in_sizes; (void)n_in; (void)out_size; (void)ws_size;
  const float* x_protein = (const float*)d_in[0];
  const float* x_class   = (const float*)d_in[1];
  const int*   src       = (const int*)d_in[2];
  const int*   dst       = (const int*)d_in[3];
  const int*   els       = (const int*)d_in[4];
  const int*   eld       = (const int*)d_in[5];
  const float* sage_pc_Wl = (const float*)d_in[6];
  const float* sage_pc_bl = (const float*)d_in[7];
  const float* sage_pc_Wr = (const float*)d_in[8];
  const float* sage_cp_Wl = (const float*)d_in[9];
  const float* sage_cp_bl = (const float*)d_in[10];
  const float* sage_cp_Wr = (const float*)d_in[11];
  const float* gat_pc_Wsrc = (const float*)d_in[12];
  const float* gat_pc_Wdst = (const float*)d_in[13];
  const float* gat_pc_asrc = (const float*)d_in[14];
  const float* gat_pc_adst = (const float*)d_in[15];
  const float* gat_pc_b    = (const float*)d_in[16];
  const float* gat_cp_Wsrc = (const float*)d_in[17];
  const float* gat_cp_Wdst = (const float*)d_in[18];
  const float* gat_cp_asrc = (const float*)d_in[19];
  const float* gat_cp_adst = (const float*)d_in[20];
  const float* gat_cp_b    = (const float*)d_in[21];
  const float* lin_p_W = (const float*)d_in[22];
  const float* lin_p_b = (const float*)d_in[23];
  const float* lin_c_W = (const float*)d_in[24];
  const float* lin_c_b = (const float*)d_in[25];
  const float* link_W  = (const float*)d_in[26];
  const float* link_b  = (const float*)d_in[27];
  float* out = (float*)d_out;

  // ---------------- workspace carve ----------------
  char* p = (char*)d_ws;
  float* y1     = bump(p, (size_t)NP * HID);
  float* y2     = bump(p, (size_t)NC * HID);
  float* agg_c  = bump(p, (size_t)NC * HID);
  float* cnt_c  = bump(p, NC);
  float* agg_p  = bump(p, (size_t)NP * HID);
  float* cnt_p  = bump(p, NP);
  float* h_c    = bump(p, (size_t)NC * HID);
  float* h_p    = bump(p, (size_t)NP * HID);
  float* hs_pc  = bump(p, (size_t)NP * HEADS * HID);
  float* hd_pc  = bump(p, (size_t)NC * HEADS * HID);
  float* hs_cp  = bump(p, (size_t)NC * HEADS * HID);
  float* hd_cp  = bump(p, (size_t)NP * HEADS * HID);
  float* als_pc = bump(p, (size_t)NP * HEADS);
  float* ald_pc = bump(p, (size_t)NC * HEADS);
  unsigned* emax_c = (unsigned*)bump(p, (size_t)NC * HEADS);
  float* den_c  = bump(p, (size_t)NC * HEADS);
  float* att_c  = bump(p, (size_t)NC * HEADS * HID);
  float* als_cp = bump(p, (size_t)NC * HEADS);
  float* ald_cp = bump(p, (size_t)NP * HEADS);
  unsigned* emax_p = (unsigned*)bump(p, (size_t)NP * HEADS);
  float* den_p  = bump(p, (size_t)NP * HEADS);
  float* att_p  = bump(p, (size_t)NP * HEADS * HID);
  float* g_c    = bump(p, (size_t)NC * HID);
  float* g_p    = bump(p, (size_t)NP * HID);
  float* vp     = bump(p, HID);
  float* vc     = bump(p, HID);
  float* s0     = bump(p, 1);
  float* sp     = bump(p, NP);
  float* sc     = bump(p, NC);

  const unsigned ENC_NEG_MAX = 0x00800000u;  // fenc(-FLT_MAX)

  // ---- SAGE layer (project-first, then scatter-mean) ----
  launch_gemm(0, x_protein, sage_pc_Wl, nullptr, nullptr, y1, NP, HID, DP, stream);
  launch_gemm(0, x_class,   sage_cp_Wl, nullptr, nullptr, y2, NC, HID, DC, stream);

  hipMemsetAsync(agg_c, 0, (size_t)NC * HID * 4, stream);
  hipMemsetAsync(cnt_c, 0, (size_t)NC * 4, stream);
  hipMemsetAsync(agg_p, 0, (size_t)NP * HID * 4, stream);
  hipMemsetAsync(cnt_p, 0, (size_t)NP * 4, stream);

  sage_scatter_kernel<<<(NE + 7) / 8, 256, 0, stream>>>(y1, src, dst, agg_c, cnt_c, NE);
  sage_scatter_kernel<<<(NE + 7) / 8, 256, 0, stream>>>(y2, dst, src, agg_p, cnt_p, NE);
  div_cnt_kernel<<<(NC * HID + 255) / 256, 256, 0, stream>>>(agg_c, cnt_c, NC);
  div_cnt_kernel<<<(NP * HID + 255) / 256, 256, 0, stream>>>(agg_p, cnt_p, NP);

  launch_gemm(7, x_class,   sage_pc_Wr, agg_c, sage_pc_bl, h_c, NC, HID, DC, stream);
  launch_gemm(7, x_protein, sage_cp_Wr, agg_p, sage_cp_bl, h_p, NP, HID, DP, stream);

  // ---- GAT projections ----
  launch_gemm(0, h_p, gat_pc_Wsrc, nullptr, nullptr, hs_pc, NP, HEADS * HID, HID, stream);
  launch_gemm(0, h_c, gat_pc_Wdst, nullptr, nullptr, hd_pc, NC, HEADS * HID, HID, stream);
  launch_gemm(0, h_c, gat_cp_Wsrc, nullptr, nullptr, hs_cp, NC, HEADS * HID, HID, stream);
  launch_gemm(0, h_p, gat_cp_Wdst, nullptr, nullptr, hd_cp, NP, HEADS * HID, HID, stream);

  attn_logits_kernel<<<(NP + 7) / 8, 256, 0, stream>>>(hs_pc, gat_pc_asrc, als_pc, NP);
  attn_logits_kernel<<<(NC + 7) / 8, 256, 0, stream>>>(hd_pc, gat_pc_adst, ald_pc, NC);
  attn_logits_kernel<<<(NC + 7) / 8, 256, 0, stream>>>(hs_cp, gat_cp_asrc, als_cp, NC);
  attn_logits_kernel<<<(NP + 7) / 8, 256, 0, stream>>>(hd_cp, gat_cp_adst, ald_cp, NP);

  fill_u32_kernel<<<(NC * HEADS + 255) / 256, 256, 0, stream>>>(emax_c, ENC_NEG_MAX, NC * HEADS);
  fill_u32_kernel<<<(NP * HEADS + 255) / 256, 256, 0, stream>>>(emax_p, ENC_NEG_MAX, NP * HEADS);
  hipMemsetAsync(den_c, 0, (size_t)NC * HEADS * 4, stream);
  hipMemsetAsync(den_p, 0, (size_t)NP * HEADS * 4, stream);
  hipMemsetAsync(att_c, 0, (size_t)NC * HEADS * HID * 4, stream);
  hipMemsetAsync(att_p, 0, (size_t)NP * HEADS * HID * 4, stream);

  // GAT p->c (segment over class nodes)
  edge_max_kernel<<<(NE + 255) / 256, 256, 0, stream>>>(als_pc, ald_pc, src, dst, emax_c, NE);
  edge_den_kernel<<<(NE + 255) / 256, 256, 0, stream>>>(als_pc, ald_pc, src, dst, emax_c, den_c, NE);
  edge_scatter_kernel<<<(NE + 7) / 8, 256, 0, stream>>>(als_pc, ald_pc, emax_c, den_c,
                                                        hs_pc, src, dst, att_c, NE);
  // GAT c->p (segment over protein nodes)
  edge_max_kernel<<<(NE + 255) / 256, 256, 0, stream>>>(als_cp, ald_cp, dst, src, emax_p, NE);
  edge_den_kernel<<<(NE + 255) / 256, 256, 0, stream>>>(als_cp, ald_cp, dst, src, emax_p, den_p, NE);
  edge_scatter_kernel<<<(NE + 7) / 8, 256, 0, stream>>>(als_cp, ald_cp, emax_p, den_p,
                                                        hs_cp, dst, src, att_p, NE);

  head_mean_kernel<<<(NC * HID + 255) / 256, 256, 0, stream>>>(att_c, gat_pc_b, g_c, NC);
  head_mean_kernel<<<(NP * HID + 255) / 256, 256, 0, stream>>>(att_p, gat_cp_b, g_p, NP);

  // ---- collapsed output projection + link predictor ----
  combine_vec_kernel<<<1, HID, 0, stream>>>(lin_p_W, lin_p_b, lin_c_W, lin_c_b,
                                            link_W, link_b, vp, vc, s0);
  node_score_kernel<<<(NP + 7) / 8, 256, 0, stream>>>(g_p, vp, sp, NP);
  node_score_kernel<<<(NC + 7) / 8, 256, 0, stream>>>(g_c, vc, sc, NC);
  link_kernel<<<(NEL + 255) / 256, 256, 0, stream>>>(sp, sc, s0, els, eld, out, NEL);
}
